// MemoryUnit_39883066310819
// MI455X (gfx1250) — compile-verified, matrix operand order checked by reading, not run on hardware
//
#include <hip/hip_runtime.h>

// ---------------------------------------------------------------------------
// MemoryUnit fused pipeline for MI455X (gfx1250, wave32, WMMA bf16,
// double-buffered async-to-LDS DMA staging).
//   N=16384, M=2000 (padded 2048), F=512
//   d_out = [ output f32 (N*F) | att f32 (N*M) ]
//   d_ws  = [ Wb bf16 (2048*512) | Wt bf16 (512*2048) | Xb bf16 (16384*512) ]
//           = 20 MB
// ---------------------------------------------------------------------------

typedef __attribute__((ext_vector_type(16))) __bf16 v16bf;
typedef __attribute__((ext_vector_type(8)))  __bf16 v8bf;
typedef __attribute__((ext_vector_type(4)))  __bf16 v4bf;
typedef __attribute__((ext_vector_type(8)))  float  v8f;

#define N_ROWS 16384
#define M_COLS 2000
#define M_PAD  2048
#define F_DIM  512

#define BM  128   // block tile rows
#define BN  128   // block tile cols
#define KC  32    // k-step (bf16 WMMA K)
#define LDT 40    // LDS row stride in bf16 elems (80B: 16B-aligned, bank-spread)

// ---- gfx1250 async DMA: global -> LDS, 16B per lane, ASYNCcnt-tracked ------
__device__ __forceinline__ void async_copy_b128(unsigned lds_off, const void* gptr) {
#if defined(__HIP_DEVICE_COMPILE__)
  asm volatile("global_load_async_to_lds_b128 %0, %1, off"
               :: "v"(lds_off), "v"(gptr)
               : "memory");
#endif
}
__device__ __forceinline__ void wait_asynccnt0() {
#if defined(__HIP_DEVICE_COMPILE__)
  asm volatile("s_wait_asynccnt 0x0" ::: "memory");
#endif
}
__device__ __forceinline__ unsigned lds_offset(const void* p) {
  // flat shared-aperture address: low 32 bits are the LDS byte offset
  return (unsigned)(size_t)p;
}

// ---- prep: f32 weight -> bf16 W (padded rows) and bf16 W^T ------------------
__global__ __launch_bounds__(256) void prep_weights(const float* __restrict__ W,
                                                    __bf16* __restrict__ Wb,
                                                    __bf16* __restrict__ Wt) {
  int idx = blockIdx.x * 256 + threadIdx.x;   // 0 .. 2048*512-1
  int m = idx >> 9;        // 0..2047
  int f = idx & 511;
  float w = (m < M_COLS) ? W[(size_t)m * F_DIM + f] : 0.0f;
  __bf16 h = (__bf16)w;
  Wb[idx] = h;                       // [2048][512]
  Wt[(size_t)f * M_PAD + m] = h;     // [512][2048]
}

// ---- prep: f32 X -> bf16 X (contiguous) ------------------------------------
__global__ __launch_bounds__(256) void prep_x(const float* __restrict__ X,
                                              __bf16* __restrict__ Xb) {
  int idx = (blockIdx.x * 256 + threadIdx.x) * 4;   // N*F multiple of 1024
  float4 v = *(const float4*)(X + idx);
  v4bf h;
  h[0] = (__bf16)v.x; h[1] = (__bf16)v.y; h[2] = (__bf16)v.z; h[3] = (__bf16)v.w;
  *(v4bf*)(Xb + idx) = h;
}

// ---- shared tiled WMMA GEMM -------------------------------------------------
// ABF16=true : A is bf16 row-major (lda), staged via async DMA (K always full)
// ABF16=false: A is f32 row-major (lda), converted on the fly; K valid to ksrc
// B: bf16 row-major (ldb), row r = K-vector of output column r, async DMA
// C: f32 row-major (ldc), columns predicated by colMax
template <bool ABF16>
__global__ __launch_bounds__(256) void gemm_bf16_wmma(
    const void* __restrict__ Av, int lda, int ksrc,
    const __bf16* __restrict__ B, int ldb,
    float* __restrict__ C, int ldc, int colMax, int ksteps) {

  __shared__ alignas(16) __bf16 sA[2][BM * LDT];
  __shared__ alignas(16) __bf16 sB[2][BN * LDT];

  const int tid  = threadIdx.x;
  const int lane = tid & 31;
  const int wave = tid >> 5;
  const int rowBase = blockIdx.x * BM;
  const int colBase = blockIdx.y * BN;
  const int wr = (wave >> 1) * 32;   // wave row offset  (4 wave-rows)
  const int wc = (wave & 1) * 64;    // wave col offset  (2 wave-cols)

  v8f acc[2][4] = {};

  const int half = lane >> 4;        // fragment K-half select per ISA layout
  const int fr   = lane & 15;        // fragment row (M for A, N for B)
  const int koff = half * 8;

  // per-thread segment coords for 16B staging (128 rows x 32 bf16 = 512 segs)
  const int segR0 = (tid + 0)   >> 2, segC0 = ((tid + 0)   & 3) * 8;
  const int segR1 = (tid + 256) >> 2, segC1 = ((tid + 256) & 3) * 8;

  // stage B tile at k0 into dst (async DMA, 2 x b128 per thread)
  auto stageB = [&](int k0, __bf16* dst) {
    async_copy_b128(lds_offset(dst + segR0 * LDT + segC0),
                    B + (size_t)(colBase + segR0) * ldb + k0 + segC0);
    async_copy_b128(lds_offset(dst + segR1 * LDT + segC1),
                    B + (size_t)(colBase + segR1) * ldb + k0 + segC1);
  };

  // stage A tile at k0 into dst
  auto stageA = [&](int k0, __bf16* dst) {
    if constexpr (ABF16) {
      const __bf16* A = (const __bf16*)Av;
      async_copy_b128(lds_offset(dst + segR0 * LDT + segC0),
                      A + (size_t)(rowBase + segR0) * lda + k0 + segC0);
      async_copy_b128(lds_offset(dst + segR1 * LDT + segC1),
                      A + (size_t)(rowBase + segR1) * lda + k0 + segC1);
    } else {
      const float* A = (const float*)Av;
      if (k0 + KC <= ksrc) {
        // hot path: block-uniform, guard-free float4 convert
#pragma unroll
        for (int i = 0; i < 4; ++i) {
          int seg = tid + i * 256;                 // 0..1023
          int r   = seg >> 3;
          int c4  = (seg & 7) * 4;
          float4 v = *(const float4*)(A + (size_t)(rowBase + r) * lda + k0 + c4);
          v4bf h;
          h[0] = (__bf16)v.x; h[1] = (__bf16)v.y; h[2] = (__bf16)v.z; h[3] = (__bf16)v.w;
          *(v4bf*)(dst + r * LDT + c4) = h;
        }
      } else {
        // K tail (at most 2 of 64 steps in GEMM2)
#pragma unroll
        for (int i = 0; i < 4; ++i) {
          int seg = tid + i * 256;
          int r   = seg >> 3;
          int c4  = (seg & 7) * 4;
          int kg  = k0 + c4;
          const float* src = A + (size_t)(rowBase + r) * lda + kg;
          v4bf h;
#pragma unroll
          for (int j = 0; j < 4; ++j)
            h[j] = (__bf16)((kg + j < ksrc) ? src[j] : 0.0f);
          *(v4bf*)(dst + r * LDT + c4) = h;
        }
      }
    }
  };

  // --- prologue: stage tile 0 into buffer 0
  stageA(0, sA[0]);
  stageB(0, sB[0]);
  wait_asynccnt0();
  __syncthreads();

  // --- main loop: DMA tile ks+1 overlaps WMMA on tile ks
  for (int ks = 0; ks < ksteps; ++ks) {
    const int cur = ks & 1;

    if (ks + 1 < ksteps) {
      stageA((ks + 1) * KC, sA[cur ^ 1]);
      stageB((ks + 1) * KC, sB[cur ^ 1]);
    }

    // build fragments per ISA 16-bit A-layout and issue 2x4 WMMAs
    v16bf af[2], bf[4];
#pragma unroll
    for (int t = 0; t < 2; ++t) {
      const __bf16* p = &sA[cur][(wr + t * 16 + fr) * LDT];
      v8bf lo = *(const v8bf*)(p + koff);
      v8bf hi = *(const v8bf*)(p + 16 + koff);
      af[t] = __builtin_shufflevector(lo, hi, 0,1,2,3,4,5,6,7,8,9,10,11,12,13,14,15);
    }
#pragma unroll
    for (int t = 0; t < 4; ++t) {
      const __bf16* p = &sB[cur][(wc + t * 16 + fr) * LDT];
      v8bf lo = *(const v8bf*)(p + koff);
      v8bf hi = *(const v8bf*)(p + 16 + koff);
      bf[t] = __builtin_shufflevector(lo, hi, 0,1,2,3,4,5,6,7,8,9,10,11,12,13,14,15);
    }
#pragma unroll
    for (int i = 0; i < 2; ++i)
#pragma unroll
      for (int j = 0; j < 4; ++j)
        acc[i][j] = __builtin_amdgcn_wmma_f32_16x16x32_bf16(
            false, af[i], false, bf[j], (short)0, acc[i][j], false, false);

    wait_asynccnt0();   // next tile resident (issued before compute)
    __syncthreads();    // all waves done reading cur / writing next
  }

  // --- store C per f32 C/D layout: VGPR r -> M=r (lanes 0-15) / M=r+8 (16-31)
  const int rowAdd = (lane >> 4) * 8;
#pragma unroll
  for (int i = 0; i < 2; ++i) {
#pragma unroll
    for (int j = 0; j < 4; ++j) {
      int tileR = rowBase + wr + i * 16;
      int tileC = colBase + wc + j * 16;
      int col = tileC + (lane & 15);
      if (col < colMax) {
#pragma unroll
        for (int r = 0; r < 8; ++r)
          C[(size_t)(tileR + rowAdd + r) * ldc + col] = acc[i][j][r];
      }
    }
  }
}

// ---- softmax + soft-shrink + L1 normalize, one block per row, in place -----
__global__ __launch_bounds__(256) void softshrink_norm(float* __restrict__ att) {
  const int M = M_COLS;
  float* p = att + (size_t)blockIdx.x * M;
  const int tid = threadIdx.x, lane = tid & 31, wave = tid >> 5;
  __shared__ float red[8];

  float v[8];
#pragma unroll
  for (int t = 0; t < 8; ++t) {
    int i = tid + t * 256;
    v[t] = (i < M) ? p[i] : -3.402823466e38f;
  }

  // block max
  float mx = v[0];
#pragma unroll
  for (int t = 1; t < 8; ++t) mx = fmaxf(mx, v[t]);
#pragma unroll
  for (int off = 16; off >= 1; off >>= 1) mx = fmaxf(mx, __shfl_xor(mx, off, 32));
  if (lane == 0) red[wave] = mx;
  __syncthreads();
  float bm = red[0];
#pragma unroll
  for (int w = 1; w < 8; ++w) bm = fmaxf(bm, red[w]);
  __syncthreads();

  // exp + block sum
  float e[8]; float s = 0.0f;
#pragma unroll
  for (int t = 0; t < 8; ++t) {
    int i = tid + t * 256;
    e[t] = (i < M) ? __expf(v[t] - bm) : 0.0f;
    s += e[t];
  }
#pragma unroll
  for (int off = 16; off >= 1; off >>= 1) s += __shfl_xor(s, off, 32);
  if (lane == 0) red[wave] = s;
  __syncthreads();
  float bs = 0.0f;
#pragma unroll
  for (int w = 0; w < 8; ++w) bs += red[w];
  __syncthreads();

  // shrink + L1
  const float inv = 1.0f / bs;
  float a[8]; float l1 = 0.0f;
#pragma unroll
  for (int t = 0; t < 8; ++t) {
    float pr = e[t] * inv;
    float sh = pr - 0.05f;
    a[t] = (sh > 0.0f) ? pr * sh / (sh + 1e-12f) : 0.0f;
    l1 += a[t];
  }
#pragma unroll
  for (int off = 16; off >= 1; off >>= 1) l1 += __shfl_xor(l1, off, 32);
  if (lane == 0) red[wave] = l1;
  __syncthreads();
  float bl = 0.0f;
#pragma unroll
  for (int w = 0; w < 8; ++w) bl += red[w];

  const float nrm = 1.0f / fmaxf(bl, 1e-12f);
#pragma unroll
  for (int t = 0; t < 8; ++t) {
    int i = tid + t * 256;
    if (i < M) p[i] = a[t] * nrm;
  }
}

// ---------------------------------------------------------------------------
extern "C" void kernel_launch(void* const* d_in, const int* in_sizes, int n_in,
                              void* d_out, int out_size, void* d_ws, size_t ws_size,
                              hipStream_t stream) {
  const float* X = (const float*)d_in[0];   // [16384, 512]
  const float* W = (const float*)d_in[1];   // [2000, 512]
  float* out = (float*)d_out;                          // [16384, 512]
  float* att = out + (size_t)N_ROWS * F_DIM;           // [16384, 2000]
  __bf16* Wb = (__bf16*)d_ws;                          // [2048, 512]
  __bf16* Wt = Wb + (size_t)M_PAD * F_DIM;             // [512, 2048]
  __bf16* Xb = Wt + (size_t)F_DIM * M_PAD;             // [16384, 512]

  // 1) bf16 weight (+transpose) and bf16 X
  prep_weights<<<(M_PAD * F_DIM) / 256, 256, 0, stream>>>(W, Wb, Wt);
  prep_x<<<(N_ROWS * F_DIM) / (256 * 4), 256, 0, stream>>>(X, Xb);

  // 2) logits = X @ W^T  -> att region (raw), K=512, fully async-DMA staged
  gemm_bf16_wmma<true><<<dim3(N_ROWS / BM, 16), 256, 0, stream>>>(
      Xb, F_DIM, F_DIM, Wb, F_DIM, att, M_COLS, M_COLS, F_DIM / KC);

  // 3) softmax + shrink + L1-normalize in place
  softshrink_norm<<<N_ROWS, 256, 0, stream>>>(att);

  // 4) out = att @ W  (B = W^T rows), K padded to 2048 with zeros
  gemm_bf16_wmma<false><<<dim3(N_ROWS / BM, F_DIM / BN), 256, 0, stream>>>(
      att, M_COLS, M_COLS, Wt, M_PAD, out, F_DIM, F_DIM, M_PAD / KC);
}
